// Encoder_75058848465262
// MI455X (gfx1250) — matrix operand order repaired and checked by reading, not compiled, not run on hardware
//
#include <hip/hip_runtime.h>

typedef __bf16 bf16_t;
typedef __attribute__((ext_vector_type(16))) __bf16 v16bf;
typedef __attribute__((ext_vector_type(8)))  __bf16 v8bf;
typedef __attribute__((ext_vector_type(8)))  float   v8f;

namespace {
constexpr int B_   = 64;     // batch
constexpr int S_   = 256;    // seq len
constexpr int DCx  = 100;    // char emb dim (bichar same)
constexpr int DINx = 400;    // concat dim
constexpr int KCx  = 416;    // concat dim padded to 13*32
constexpr int DRNNx= 200;    // linear out dim
constexpr int NPx  = 224;    // linear out padded (7*32) = x-part of LSTM K
constexpr int Hx   = 256;    // LSTM hidden
constexpr int KLx  = 480;    // NPx + Hx = 15*32
constexpr int G4x  = 1024;   // 4*H gate dim
}

__device__ __forceinline__ float sigmf(float x) { return 1.0f / (1.0f + __expf(-x)); }

// A fragment: 16-bit A 16x32 layout (ISA 7.12.2). lane = 16*half + m.
// elements 0..7  <-> K = k0 + 8*half      .. +8
// elements 8..15 <-> K = k0 + 16 + 8*half .. +8
__device__ __forceinline__ v16bf fragA(const bf16_t* __restrict__ row, int k0, int half) {
  v8bf lo = *reinterpret_cast<const v8bf*>(row + k0 + 8 * half);
  v8bf hi = *reinterpret_cast<const v8bf*>(row + k0 + 16 + 8 * half);
  return __builtin_shufflevector(lo, hi, 0,1,2,3,4,5,6,7,8,9,10,11,12,13,14,15);
}

// B fragment: lane = 16*half + n, elements 0..7 <-> K = k0 + 8*half .. +8, 8..15 <-> K = k0 + 16 + 8*half .. +8
// With gates = X * W^T and W row-major [N][K], B-tile lane reads W row (n0 + n).
__device__ __forceinline__ v16bf fragB(const bf16_t* __restrict__ wrow, int k0, int half) {
  v8bf lo = *reinterpret_cast<const v8bf*>(wrow + k0 + 8 * half);
  v8bf hi = *reinterpret_cast<const v8bf*>(wrow + k0 + 16 + 8 * half);
  return __builtin_shufflevector(lo, hi, 0,1,2,3,4,5,6,7,8,9,10,11,12,13,14,15);
}

__device__ __forceinline__ v8f wmma_bf16(v16bf a, v16bf b, v8f c) {
  return __builtin_amdgcn_wmma_f32_16x16x32_bf16(false, a, false, b, (short)0, c, false, false);
}

// ---------------------------------------------------------------------------
// Gather embeddings into bf16 concat rows, row index r = t*64 + b, K padded to 416
__global__ void k_gather(const int* __restrict__ cf,  const int* __restrict__ bl,
                         const int* __restrict__ br,  const int* __restrict__ sc,
                         const int* __restrict__ sbl, const int* __restrict__ sbr,
                         const float* __restrict__ ce,  const float* __restrict__ be,
                         const float* __restrict__ sce, const float* __restrict__ sbe,
                         bf16_t* __restrict__ catL, bf16_t* __restrict__ catR) {
  long idx = (long)blockIdx.x * blockDim.x + threadIdx.x;
  const long total = (long)B_ * S_ * KCx;
  if (idx >= total) return;
  int k = (int)(idx % KCx);
  int r = (int)(idx / KCx);
  int b = r & (B_ - 1), t = r >> 6;
  int p = b * S_ + t;                       // index arrays are [B,S]
  float vL = 0.f, vR = 0.f;
  if (k < DCx) {
    float v = ce[(long)cf[p] * DCx + k];                 vL = v; vR = v;
  } else if (k < 200) {
    float v = sce[(long)sc[p] * DCx + (k - 100)];        vL = v; vR = v;
  } else if (k < 300) {
    vL = be[(long)bl[p] * DCx + (k - 200)];
    vR = be[(long)br[p] * DCx + (k - 200)];
  } else if (k < DINx) {
    vL = sbe[(long)sbl[p] * DCx + (k - 300)];
    vR = sbe[(long)sbr[p] * DCx + (k - 300)];
  }
  catL[idx] = (bf16_t)vL;
  catR[idx] = (bf16_t)vR;
}

// W_lin [200,400] f32 -> bf16 [224,416] zero-padded; bias padded to 224 (tanh(0)=0 pads)
__global__ void k_pack_wlin(const float* __restrict__ W, const float* __restrict__ bias,
                            bf16_t* __restrict__ Wp, float* __restrict__ bp) {
  int idx = blockIdx.x * blockDim.x + threadIdx.x;
  if (idx >= NPx * KCx) return;
  int k = idx % KCx, n = idx / KCx;
  float v = (n < DRNNx && k < DINx) ? W[n * DINx + k] : 0.f;
  Wp[idx] = (bf16_t)v;
  if (k == 0) bp[n] = (n < DRNNx) ? bias[n] : 0.f;
}

// Wcat [1024,480] bf16 = [Wih | zeros(24) | Whh]; bcat = bih + bhh
__global__ void k_pack_wcat(const float* __restrict__ Wih, const float* __restrict__ Whh,
                            const float* __restrict__ bih, const float* __restrict__ bhh,
                            bf16_t* __restrict__ Wc, float* __restrict__ bc) {
  int idx = blockIdx.x * blockDim.x + threadIdx.x;
  if (idx >= G4x * KLx) return;
  int k = idx % KLx, n = idx / KLx;
  float v;
  if (k < DRNNx)      v = Wih[n * DRNNx + k];
  else if (k < NPx)   v = 0.f;
  else                v = Whh[n * Hx + (k - NPx)];
  Wc[idx] = (bf16_t)v;
  if (k == 0) bc[n] = bih[n] + bhh[n];
}

// ---------------------------------------------------------------------------
// Phase-1 GEMM: xpad = tanh(cat @ Wlin^T + b), M=16384, K=416, N=224, bf16 WMMA
__global__ void k_lin(const bf16_t* __restrict__ catL, const bf16_t* __restrict__ catR,
                      const bf16_t* __restrict__ Wp, const float* __restrict__ bp,
                      bf16_t* __restrict__ xpadL, bf16_t* __restrict__ xpadR) {
  int wid  = (blockIdx.x * blockDim.x + threadIdx.x) >> 5;
  int lane = threadIdx.x & 31;
  const int MT = (B_ * S_) / 16;   // 1024
  const int NT = NPx / 16;         // 14
  if (wid >= 2 * MT * NT) return;
  int side = wid / (MT * NT);
  int rem  = wid % (MT * NT);
  int mt = rem / NT, nt = rem % NT;
  const bf16_t* A = side ? catR : catL;
  bf16_t*       O = side ? xpadR : xpadL;
  int half = lane >> 4, lm = lane & 15;
  const bf16_t* arow = A  + (size_t)(mt * 16 + lm) * KCx;
  const bf16_t* brow = Wp + (size_t)(nt * 16 + lm) * KCx;
  v8f acc = {};
#pragma unroll
  for (int k0 = 0; k0 < KCx; k0 += 32)
    acc = wmma_bf16(fragA(arow, k0, half), fragB(brow, k0, half), acc);
  int col = nt * 16 + lm;
  float bias = bp[col];
#pragma unroll
  for (int r = 0; r < 8; ++r) {
    int row = mt * 16 + r + 8 * half;           // C/D layout: M = r + 8*half, N = lane&15
    O[(size_t)row * NPx + col] = (bf16_t)tanhf(acc[r] + bias);
  }
}

// ---------------------------------------------------------------------------
// One LSTM time step, both directions. 128 waves: dir(2) x mtile(4) x ngroup(16).
// Each wave computes i/f/g/o tiles for its (rows, hidden-cols) and does the cell
// update locally -> no inter-block sync. h double-buffered across steps.
__global__ void k_step(const bf16_t* __restrict__ xpadL, const bf16_t* __restrict__ xpadR,
                       const bf16_t* __restrict__ WcL,   const bf16_t* __restrict__ WcR,
                       const float* __restrict__ bcL,    const float* __restrict__ bcR,
                       bf16_t* __restrict__ hbuf, float* __restrict__ cbuf,
                       float* __restrict__ out, int s) {
  int wid  = (blockIdx.x * blockDim.x + threadIdx.x) >> 5;
  int lane = threadIdx.x & 31;
  if (wid >= 2 * 4 * 16) return;
  int dir = wid >> 6;
  int mt  = (wid >> 4) & 3;
  int ng  = wid & 15;
  int half = lane >> 4, lm = lane & 15;
  int t = dir ? (S_ - 1 - s) : s;

  const bf16_t* xp = dir ? xpadR : xpadL;
  const bf16_t* Wc = dir ? WcR : WcL;
  const float*  bc = dir ? bcR : bcL;
  const bf16_t* hprev = hbuf + (size_t)(dir * 2 + (s & 1)) * (B_ * Hx);
  bf16_t*       hnext = hbuf + (size_t)(dir * 2 + ((s + 1) & 1)) * (B_ * Hx);
  float*        cst   = cbuf + (size_t)dir * (B_ * Hx);

  int arow = mt * 16 + lm;
  const bf16_t* xrow = xp + ((size_t)t * B_ + arow) * NPx;
  const bf16_t* hrow = hprev + (size_t)arow * Hx;
  const bf16_t* w0 = Wc + (size_t)(0 * Hx + ng * 16 + lm) * KLx;   // i
  const bf16_t* w1 = Wc + (size_t)(1 * Hx + ng * 16 + lm) * KLx;   // f
  const bf16_t* w2 = Wc + (size_t)(2 * Hx + ng * 16 + lm) * KLx;   // g
  const bf16_t* w3 = Wc + (size_t)(3 * Hx + ng * 16 + lm) * KLx;   // o

  v8f ci = {}, cfv = {}, cg = {}, co = {};
#pragma unroll
  for (int k0 = 0; k0 < KLx; k0 += 32) {
    v16bf a = (k0 < NPx) ? fragA(xrow, k0, half) : fragA(hrow, k0 - NPx, half);
    ci  = wmma_bf16(a, fragB(w0, k0, half), ci);
    cfv = wmma_bf16(a, fragB(w1, k0, half), cfv);
    cg  = wmma_bf16(a, fragB(w2, k0, half), cg);
    co  = wmma_bf16(a, fragB(w3, k0, half), co);
  }

  int col = ng * 16 + lm;
  float bi = bc[col], bf_ = bc[Hx + col], bg = bc[2 * Hx + col], bo = bc[3 * Hx + col];
#pragma unroll
  for (int r = 0; r < 8; ++r) {
    int row = mt * 16 + r + 8 * half;
    float iv = sigmf(ci[r]  + bi);
    float fv = sigmf(cfv[r] + bf_);
    float gv = tanhf(cg[r]  + bg);
    float ov = sigmf(co[r]  + bo);
    size_t cidx = (size_t)row * Hx + col;
    float cn = fv * cst[cidx] + iv * gv;
    cst[cidx] = cn;
    float hv = ov * tanhf(cn);
    hnext[cidx] = (bf16_t)hv;
    out[((size_t)row * S_ + t) * (2 * Hx) + (size_t)dir * Hx + col] = hv;  // [B,S,512]
  }
}

// ---------------------------------------------------------------------------
extern "C" void kernel_launch(void* const* d_in, const int* in_sizes, int n_in,
                              void* d_out, int out_size, void* d_ws, size_t ws_size,
                              hipStream_t stream) {
  (void)in_sizes; (void)n_in; (void)out_size; (void)ws_size;
  const int*   cf  = (const int*)d_in[0];
  const int*   bl  = (const int*)d_in[1];
  const int*   br  = (const int*)d_in[2];
  const int*   sc  = (const int*)d_in[3];
  const int*   sbl = (const int*)d_in[4];
  const int*   sbr = (const int*)d_in[5];
  const float* ce  = (const float*)d_in[6];
  const float* be  = (const float*)d_in[7];
  const float* sce = (const float*)d_in[8];
  const float* sbe = (const float*)d_in[9];
  const float* Wlin = (const float*)d_in[10];
  const float* blin = (const float*)d_in[11];
  const float* WihL = (const float*)d_in[12];
  const float* WhhL = (const float*)d_in[13];
  const float* bihL = (const float*)d_in[14];
  const float* bhhL = (const float*)d_in[15];
  const float* WihR = (const float*)d_in[16];
  const float* WhhR = (const float*)d_in[17];
  const float* bihR = (const float*)d_in[18];
  const float* bhhR = (const float*)d_in[19];
  float* out = (float*)d_out;

  char* ws = (char*)d_ws;
  size_t off = 0;
  auto carve = [&](size_t bytes) -> void* {
    void* p = ws + off;
    off = (off + bytes + 255) & ~(size_t)255;
    return p;
  };
  const size_t ROWS = (size_t)B_ * S_;                 // 16384
  bf16_t* catL  = (bf16_t*)carve(ROWS * KCx * 2);      // 13.6 MB
  bf16_t* catR  = (bf16_t*)carve(ROWS * KCx * 2);
  bf16_t* Wp    = (bf16_t*)carve((size_t)NPx * KCx * 2);
  float*  bp    = (float*) carve((size_t)NPx * 4);
  bf16_t* xpadL = (bf16_t*)carve(ROWS * NPx * 2);      // 7.3 MB
  bf16_t* xpadR = (bf16_t*)carve(ROWS * NPx * 2);
  bf16_t* WcL   = (bf16_t*)carve((size_t)G4x * KLx * 2);
  bf16_t* WcR   = (bf16_t*)carve((size_t)G4x * KLx * 2);
  float*  bcLp  = (float*) carve((size_t)G4x * 4);
  float*  bcRp  = (float*) carve((size_t)G4x * 4);
  bf16_t* hbuf  = (bf16_t*)carve((size_t)2 * 2 * B_ * Hx * 2);   // dir x slot x B x H
  float*  cbuf  = (float*) carve((size_t)2 * B_ * Hx * 4);

  // Zero recurrent state every call (graph-capture safe).
  hipMemsetAsync(hbuf, 0, (size_t)2 * 2 * B_ * Hx * 2, stream);
  hipMemsetAsync(cbuf, 0, (size_t)2 * B_ * Hx * 4, stream);

  // Prep
  {
    long total = (long)B_ * S_ * KCx;
    k_gather<<<(int)((total + 255) / 256), 256, 0, stream>>>(
        cf, bl, br, sc, sbl, sbr, ce, be, sce, sbe, catL, catR);
  }
  k_pack_wlin<<<(NPx * KCx + 255) / 256, 256, 0, stream>>>(Wlin, blin, Wp, bp);
  k_pack_wcat<<<(G4x * KLx + 255) / 256, 256, 0, stream>>>(WihL, WhhL, bihL, bhhL, WcL, bcLp);
  k_pack_wcat<<<(G4x * KLx + 255) / 256, 256, 0, stream>>>(WihR, WhhR, bihR, bhhR, WcR, bcRp);

  // Phase-1 WMMA GEMM + tanh: 2 sides x 1024 Mtiles x 14 Ntiles waves
  {
    int waves = 2 * ((B_ * S_) / 16) * (NPx / 16);     // 28672
    k_lin<<<waves * 32 / 256, 256, 0, stream>>>(catL, catR, Wp, bp, xpadL, xpadR);
  }

  // Sequential LSTM scan: 256 step launches, both directions per step.
  for (int s = 0; s < S_; ++s) {
    k_step<<<16, 256, 0, stream>>>(xpadL, xpadR, WcL, WcR, bcLp, bcRp, hbuf, cbuf, out, s);
  }
}
